// MoEBertSelfAttention_72206990180769
// MI455X (gfx1250) — matrix operand order repaired and checked by reading, not compiled, not run on hardware
//
#include <hip/hip_runtime.h>
#include <hip/hip_bf16.h>

typedef __attribute__((ext_vector_type(16))) __bf16 v16bf;
typedef __attribute__((ext_vector_type(8)))  __bf16 v8bf;
typedef __attribute__((ext_vector_type(4)))  __bf16 v4bf;
typedef __attribute__((ext_vector_type(8)))  float  v8f;

#define WMMA_BF16(A,B,C) __builtin_amdgcn_wmma_f32_16x16x32_bf16(false,(A),false,(B),(short)0,(C),false,false)

static __device__ inline v16bf cat8(v8bf lo, v8bf hi){
  v16bf r;
  #pragma unroll
  for (int i = 0; i < 8; ++i){ r[i] = lo[i]; r[i+8] = hi[i]; }
  return r;
}

// async 16B copy global -> LDS (per-lane), tracked by ASYNCcnt
static __device__ inline void async_copy16(unsigned lds_off, const void* gaddr){
  asm volatile("global_load_async_to_lds_b128 %0, %1, off"
               :: "v"(lds_off), "v"((unsigned long long)(uintptr_t)gaddr)
               : "memory");
}

// ---------------- fp32 -> bf16 convert ----------------
__global__ __launch_bounds__(256)
void cvt_f32_bf16(const float* __restrict__ src, __bf16* __restrict__ dst, int n){
  int i = (blockIdx.x * 256 + threadIdx.x) * 4;
  if (i >= n) return;
  float4 f = *(const float4*)(src + i);
  v4bf o;
  o[0] = (__bf16)f.x; o[1] = (__bf16)f.y; o[2] = (__bf16)f.z; o[3] = (__bf16)f.w;
  *(v4bf*)(dst + i) = o;
}

// ---------------- QKV projection: C[m,n] = sum_k X[m,k]*W[n,k] + b[n] ----------------
// mode 0/1: dst = [B,H,S,Dh] bf16 (Q or K).  mode 2: dst = [B,H,Dh,S] bf16 (V transposed).
__global__ __launch_bounds__(256)
void qkv_gemm(const __bf16* __restrict__ X, const __bf16* __restrict__ W,
              const float* __restrict__ bias, __bf16* __restrict__ dst, int mode){
  const int lane  = threadIdx.x & 31;
  const int wave  = threadIdx.x >> 5;
  const int tileM = blockIdx.x >> 4;          // 64 tiles of 128 rows
  const int tileN = blockIdx.x & 15;          // 16 tiles of 64 cols
  const int m0 = tileM * 128 + (wave & 3) * 32;
  const int n0 = tileN * 64  + (wave >> 2) * 32;
  const int lr    = lane & 15;
  const int abase = (lane < 16) ? 0 : 16;     // A fragment K half (contiguous, matches B)
  const int bbase = (lane < 16) ? 0 : 16;     // B fragment K half
  const int rowhi = (lane >> 4) * 8;

  v8f acc[2][2] = {};
  for (int k0 = 0; k0 < 1024; k0 += 32){
    v16bf a[2], b[2];
    #pragma unroll
    for (int i = 0; i < 2; ++i){
      const __bf16* p = X + (size_t)(m0 + i*16 + lr) * 1024 + k0 + abase;
      a[i] = cat8(*(const v8bf*)p, *(const v8bf*)(p + 8));
    }
    #pragma unroll
    for (int j = 0; j < 2; ++j){
      const __bf16* p = W + (size_t)(n0 + j*16 + lr) * 1024 + k0 + bbase;
      b[j] = *(const v16bf*)p;
    }
    #pragma unroll
    for (int i = 0; i < 2; ++i)
      #pragma unroll
      for (int j = 0; j < 2; ++j)
        acc[i][j] = WMMA_BF16(a[i], b[j], acc[i][j]);
  }

  #pragma unroll
  for (int i = 0; i < 2; ++i){
    #pragma unroll
    for (int j = 0; j < 2; ++j){
      const int n  = n0 + j*16 + lr;
      const float bn = bias[n];
      const int h = n >> 6, d = n & 63;
      #pragma unroll
      for (int r = 0; r < 8; ++r){
        const int m = m0 + i*16 + rowhi + r;
        const int bidx = m >> 10, s = m & 1023;
        const float v = acc[i][j][r] + bn;
        const size_t off = (mode < 2)
            ? ((((size_t)bidx * 16 + h) * 1024 + s) * 64 + d)
            : ((((size_t)bidx * 16 + h) * 64 + d) * 1024 + s);
        dst[off] = (__bf16)v;
      }
    }
  }
}

// ---------------- fused flash attention ----------------
// Q,K: [B,H,S,64] bf16. Vt: [B,H,64,S] bf16. out: [B,S,H*64] f32.
// Block = 8 waves x 16 q-rows = 128 q rows. K/V tiles (64 keys) staged in LDS by
// double-buffered global_load_async_to_lds_b128 and shared by all 8 waves.
__global__ __launch_bounds__(256)
void flash_attn(const __bf16* __restrict__ Q, const __bf16* __restrict__ K,
                const __bf16* __restrict__ Vt, const float* __restrict__ mask,
                const float* __restrict__ head_mask, float* __restrict__ out){
  __shared__ __bf16 Ktile[2][64 * 64];        // [buf][key][d]   8 KB each
  __shared__ __bf16 Vtile[2][64 * 64];        // [buf][d][key]   8 KB each
  __shared__ __bf16 plds[8][16 * 64];         // per-wave P transpose, 16 KB

  const int tid  = threadIdx.x;
  const int lane = tid & 31;
  const int wave = tid >> 5;
  const int qblk = blockIdx.x & 7;            // 8 q-blocks of 128 rows
  const int bh   = blockIdx.x >> 3;           // 0..127
  const int b = bh >> 4, h = bh & 15;
  const int q0 = qblk * 128 + wave * 16;      // this wave's 16-row Q strip

  const __bf16* Qw = Q  + ((size_t)bh * 1024 + q0) * 64;
  const __bf16* Kw = K  + (size_t)bh * 1024 * 64;
  const __bf16* Vw = Vt + (size_t)bh * 64 * 1024;
  const float*  mrow = mask + (size_t)b * 1024;

  const int lr    = lane & 15;
  const int abase = (lane < 16) ? 0 : 16;
  const int bbase = (lane < 16) ? 0 : 16;
  const int rowhi = (lane >> 4) * 8;
  __bf16* Lp = plds[wave];

  const unsigned ldsK0 = (unsigned)(uintptr_t)&Ktile[0][0];
  const unsigned ldsK1 = (unsigned)(uintptr_t)&Ktile[1][0];
  const unsigned ldsV0 = (unsigned)(uintptr_t)&Vtile[0][0];
  const unsigned ldsV1 = (unsigned)(uintptr_t)&Vtile[1][0];

  // stage one 64-key block (8 KB K + 8 KB V) into LDS buffer `buf`
  auto prefetch = [&](int buf, int kb){
    const int key0 = kb * 64;
    const unsigned lk = buf ? ldsK1 : ldsK0;
    const unsigned lv = buf ? ldsV1 : ldsV0;
    const __bf16* ksrc = Kw + (size_t)key0 * 64;   // contiguous 8192 B
    #pragma unroll
    for (int c = 0; c < 2; ++c){
      const int idx = tid + c * 256;               // 0..511 chunks of 16 B
      async_copy16(lk + idx * 16, (const char*)ksrc + idx * 16);
    }
    #pragma unroll
    for (int c = 0; c < 2; ++c){
      const int idx = tid + c * 256;               // row = d, 8 chunks per row
      const int d = idx >> 3, cc = idx & 7;
      async_copy16(lv + idx * 16,
                   (const char*)(Vw + (size_t)d * 1024 + key0) + cc * 16);
    }
  };

  // Q strip as two A fragments (K = 0..31, 32..63)
  v16bf aq[2];
  #pragma unroll
  for (int c = 0; c < 2; ++c){
    const __bf16* p = Qw + (size_t)lr * 64 + c * 32 + abase;
    aq[c] = cat8(*(const v8bf*)p, *(const v8bf*)(p + 8));
  }

  v8f O[4] = {};
  v8f Mrow, Lrow;
  #pragma unroll
  for (int r = 0; r < 8; ++r){ Mrow[r] = -1e30f; Lrow[r] = 0.f; }

  prefetch(0, 0);

  for (int kb = 0; kb < 16; ++kb){
    const int cur = kb & 1;
    const int key0 = kb * 64;

    if (kb < 15){
      prefetch(cur ^ 1, kb + 1);
      asm volatile("s_wait_asynccnt 0x4" ::: "memory");  // current buffer's 4 ops done
    } else {
      asm volatile("s_wait_asynccnt 0x0" ::: "memory");
    }
    __syncthreads();                                     // all waves' copies visible

    const __bf16* Kt = Ktile[cur];
    const __bf16* Vs = Vtile[cur];

    // ---- scores: S = (Q . K^T) * 0.125 + mask ----
    v8f s[4];
    #pragma unroll
    for (int t = 0; t < 4; ++t){
      const __bf16* kp = Kt + (t * 16 + lr) * 64;        // [key][d]
      v16bf bk0 = *(const v16bf*)(kp + bbase);
      v16bf bk1 = *(const v16bf*)(kp + 32 + bbase);
      v8f acc = {};
      acc = WMMA_BF16(aq[0], bk0, acc);
      acc = WMMA_BF16(aq[1], bk1, acc);
      const float mv = mrow[key0 + t * 16 + lr];
      #pragma unroll
      for (int r = 0; r < 8; ++r) s[t][r] = acc[r] * 0.125f + mv;
    }

    // ---- online softmax over this 64-key block ----
    v8f tmax;
    #pragma unroll
    for (int r = 0; r < 8; ++r)
      tmax[r] = fmaxf(fmaxf(s[0][r], s[1][r]), fmaxf(s[2][r], s[3][r]));
    #pragma unroll
    for (int m = 1; m < 16; m <<= 1)
      #pragma unroll
      for (int r = 0; r < 8; ++r)
        tmax[r] = fmaxf(tmax[r], __shfl_xor(tmax[r], m, 32));

    v8f Mnew, alpha;
    #pragma unroll
    for (int r = 0; r < 8; ++r){
      Mnew[r]  = fmaxf(Mrow[r], tmax[r]);
      alpha[r] = __expf(Mrow[r] - Mnew[r]);
    }
    v8f psum = {};
    #pragma unroll
    for (int t = 0; t < 4; ++t)
      #pragma unroll
      for (int r = 0; r < 8; ++r){
        s[t][r] = __expf(s[t][r] - Mnew[r]);
        psum[r] += s[t][r];
      }
    #pragma unroll
    for (int m = 1; m < 16; m <<= 1)
      #pragma unroll
      for (int r = 0; r < 8; ++r)
        psum[r] += __shfl_xor(psum[r], m, 32);
    #pragma unroll
    for (int r = 0; r < 8; ++r){
      Lrow[r] = Lrow[r] * alpha[r] + psum[r];
      Mrow[r] = Mnew[r];
    }
    #pragma unroll
    for (int t = 0; t < 4; ++t)
      #pragma unroll
      for (int r = 0; r < 8; ++r)
        O[t][r] *= alpha[r];

    // ---- transpose P through LDS: C-layout -> A-fragment layout ----
    #pragma unroll
    for (int t = 0; t < 4; ++t)
      #pragma unroll
      for (int r = 0; r < 8; ++r)
        Lp[(rowhi + r) * 64 + t * 16 + lr] = (__bf16)s[t][r];
    asm volatile("s_wait_dscnt 0" ::: "memory");

    v16bf pa[2];
    #pragma unroll
    for (int c = 0; c < 2; ++c){
      const __bf16* p = Lp + lr * 64 + c * 32 + abase;
      pa[c] = cat8(*(const v8bf*)p, *(const v8bf*)(p + 8));
    }

    // ---- O += P . V  (B fragments from LDS V tile, K = key dim) ----
    #pragma unroll
    for (int t = 0; t < 4; ++t){
      const __bf16* vp = Vs + (t * 16 + lr) * 64;        // [d][key]
      v16bf bv0 = *(const v16bf*)(vp + bbase);
      v16bf bv1 = *(const v16bf*)(vp + 32 + bbase);
      O[t] = WMMA_BF16(pa[0], bv0, O[t]);
      O[t] = WMMA_BF16(pa[1], bv1, O[t]);
    }

    __syncthreads();   // all waves done with `cur` before it is overwritten
  }

  // ---- normalize, apply head_mask, write [B,S,H*Dh] ----
  const float hm = head_mask[h];
  #pragma unroll
  for (int t = 0; t < 4; ++t){
    #pragma unroll
    for (int r = 0; r < 8; ++r){
      const int row = q0 + rowhi + r;
      out[((size_t)b * 1024 + row) * 1024 + h * 64 + t * 16 + lr] =
          O[t][r] * hm / Lrow[r];
    }
  }
}

extern "C" void kernel_launch(void* const* d_in, const int* in_sizes, int n_in,
                              void* d_out, int out_size, void* d_ws, size_t ws_size,
                              hipStream_t stream) {
  const float* hs   = (const float*)d_in[0];
  const float* mask = (const float*)d_in[1];
  const float* hmsk = (const float*)d_in[2];
  const float* Wq   = (const float*)d_in[3];
  const float* bq   = (const float*)d_in[4];
  const float* Wk   = (const float*)d_in[5];
  const float* bk   = (const float*)d_in[6];
  const float* Wv   = (const float*)d_in[7];
  const float* bv   = (const float*)d_in[8];
  float* out = (float*)d_out;

  char* w = (char*)d_ws;
  __bf16* Xb  = (__bf16*)w; w += (size_t)8192 * 1024 * 2;
  __bf16* Wqb = (__bf16*)w; w += (size_t)1024 * 1024 * 2;
  __bf16* Wkb = (__bf16*)w; w += (size_t)1024 * 1024 * 2;
  __bf16* Wvb = (__bf16*)w; w += (size_t)1024 * 1024 * 2;
  __bf16* Qb  = (__bf16*)w; w += (size_t)8 * 16 * 1024 * 64 * 2;
  __bf16* Kb  = (__bf16*)w; w += (size_t)8 * 16 * 1024 * 64 * 2;
  __bf16* Vtb = (__bf16*)w; w += (size_t)8 * 16 * 1024 * 64 * 2;

  cvt_f32_bf16<<<8192, 256, 0, stream>>>(hs, Xb, 8192 * 1024);
  cvt_f32_bf16<<<1024, 256, 0, stream>>>(Wq, Wqb, 1024 * 1024);
  cvt_f32_bf16<<<1024, 256, 0, stream>>>(Wk, Wkb, 1024 * 1024);
  cvt_f32_bf16<<<1024, 256, 0, stream>>>(Wv, Wvb, 1024 * 1024);

  qkv_gemm<<<1024, 256, 0, stream>>>(Xb, Wqb, bq, Qb, 0);
  qkv_gemm<<<1024, 256, 0, stream>>>(Xb, Wkb, bk, Kb, 1);
  qkv_gemm<<<1024, 256, 0, stream>>>(Xb, Wvb, bv, Vtb, 2);

  flash_attn<<<1024, 256, 0, stream>>>(Qb, Kb, Vtb, mask, hmsk, out);
}
